// HybridMambaTransformer_82291573391916
// MI455X (gfx1250) — matrix operand order repaired and checked by reading, not compiled, not run on hardware
//
#include <hip/hip_runtime.h>
#include <hip/hip_bf16.h>

typedef __attribute__((ext_vector_type(16))) _Float16 v16h;
typedef __attribute__((ext_vector_type(8)))  _Float16 v8h;
typedef __attribute__((ext_vector_type(8)))  float    v8f;
typedef _Float16 half_t;

#define TOK    8192   // B_SZ * L_SEQ
#define BSZ    16
#define LSEQ   512
#define HDIM   256
#define DIN    512
#define DSTATE 16
#define NHEADS 8
#define DFF    1024
#define XPAD   48     // padded width for xp / w_xp ( 33 -> 48 )

// ---------------- device math helpers ----------------
__device__ __forceinline__ float sigf(float x)  { return 1.f / (1.f + __expf(-x)); }
__device__ __forceinline__ float siluf(float x) { return x * sigf(x); }
__device__ __forceinline__ float geluf(float x) { return 0.5f * x * (1.f + erff(x * 0.70710678118f)); }
__device__ __forceinline__ float softplusf_(float x) { return (x > 20.f) ? x : log1pf(__expf(x)); }

// pack 16 f32 (4x float4) -> v16h fragment
__device__ __forceinline__ v16h cvt16(float4 a, float4 b, float4 c, float4 d)
{
    v16h o;
    o[0] = (_Float16)a.x;  o[1] = (_Float16)a.y;  o[2] = (_Float16)a.z;  o[3] = (_Float16)a.w;
    o[4] = (_Float16)b.x;  o[5] = (_Float16)b.y;  o[6] = (_Float16)b.z;  o[7] = (_Float16)b.w;
    o[8] = (_Float16)c.x;  o[9] = (_Float16)c.y;  o[10] = (_Float16)c.z; o[11] = (_Float16)c.w;
    o[12] = (_Float16)d.x; o[13] = (_Float16)d.y; o[14] = (_Float16)d.z; o[15] = (_Float16)d.w;
    return o;
}

// A-fragment (16x32 f16, MxK): lane (g=lane>>4, m=lane&15)
// element e -> K = k0 + 16g + e: 16 contiguous floats (pointer already offset by 16g).
__device__ __forceinline__ v16h load_fragA(const float* Arow8g, int k0)
{
    const float4* p0 = (const float4*)(Arow8g + k0);
    const float4* p1 = (const float4*)(Arow8g + k0 + 8);
    return cvt16(p0[0], p0[1], p1[0], p1[1]);
}

// B-fragment from f16 [N,K] transposed operand: element e -> K = k0 + 16g + e,
// i.e. 16 contiguous halfs per lane = two 16-byte loads.
__device__ __forceinline__ v16h load_fragB_t(const half_t* col16g, int k0)
{
    const v8h* bp = (const v8h*)(col16g + k0);
    return __builtin_shufflevector(bp[0], bp[1],
                                   0, 1, 2, 3, 4, 5, 6, 7, 8, 9, 10, 11, 12, 13, 14, 15);
}

// ---------------- weight transpose+convert: W[K,N] f32 -> WT[N,K] f16 ----------------
__global__ void k_transpose_f16(const float* __restrict__ W, half_t* __restrict__ WT, int K, int N)
{
    int i = blockIdx.x * blockDim.x + threadIdx.x;
    if (i >= K * N) return;
    int k = i / N, c = i - k * N;
    WT[(size_t)c * K + k] = (half_t)W[i];
}

// ---------------- generic WMMA GEMM ----------------
// C[M,N] = act( A[M,K] @ B[K,N] + bias ) + res   (B given as WT[N,K] f16)
// One wave computes a 16 x (16*NS) strip. K%32==0, M%16==0, N%(16*NS)==0.
template <int NS>
__global__ void k_gemm(const float* __restrict__ A, const half_t* __restrict__ WT,
                       const float* __restrict__ bias, const float* __restrict__ res,
                       float* __restrict__ C, int M, int N, int K,
                       int lda, int ldc, int act, int ntg)
{
    int tile = blockIdx.x;
    int tn = (tile % ntg) * (16 * NS);
    int tm = (tile / ntg) << 4;
    int lane = threadIdx.x;
    int g = lane >> 4, mn = lane & 15;
    const float* Arow = A + (size_t)(tm + mn) * lda + 16 * g;
    int col0 = tn + mn;
    const half_t* Wc[NS];
#pragma unroll
    for (int s = 0; s < NS; ++s) Wc[s] = WT + (size_t)(col0 + 16 * s) * K + 16 * g;
    v8f z = {};
    v8f acc[NS];
#pragma unroll
    for (int s = 0; s < NS; ++s) acc[s] = z;
    for (int k0 = 0; k0 < K; k0 += 32) {
        v16h af = load_fragA(Arow, k0);
#pragma unroll
        for (int s = 0; s < NS; ++s) {
            v16h bf = load_fragB_t(Wc[s], k0);
            acc[s] = __builtin_amdgcn_wmma_f32_16x16x32_f16(false, af, false, bf,
                                                            (short)0, acc[s], false, false);
        }
    }
#pragma unroll
    for (int s = 0; s < NS; ++s) {
        int col = col0 + 16 * s;
        float bv = bias ? bias[col] : 0.f;
#pragma unroll
        for (int v = 0; v < 8; ++v) {
            int row = tm + v + (g << 3);
            float o = acc[s][v] + bv;
            if (act == 1) o = geluf(o);
            else if (act == 2) o = siluf(o);
            else if (act == 3) o = sigf(o);
            if (res) o += res[(size_t)row * ldc + col];
            C[(size_t)row * ldc + col] = o;
        }
    }
}

// ---------------- attention: S = scale * Q @ K^T (per head, one batch) ----------------
__global__ void k_attn_scores(const float* __restrict__ Qb, const float* __restrict__ Kb,
                              float* __restrict__ S, float scale)
{
    int h = blockIdx.z;
    int tile = blockIdx.x;                         // 32x32 tiles over 512x512
    int tn = (tile & 31) << 4, tm = (tile >> 5) << 4;
    int lane = threadIdx.x, g = lane >> 4, mn = lane & 15;
    const float* Qrow = Qb + h * 32 + (size_t)(tm + mn) * 768 + 16 * g;
    const float* Krow = Kb + h * 32 + (size_t)(tn + mn) * 768 + 16 * g;  // NT: B[k][n]=Km[n][k]
    v16h af = load_fragA(Qrow, 0);
    const float4* kp = (const float4*)Krow;        // 16 contiguous floats
    v16h bf = cvt16(kp[0], kp[1], kp[2], kp[3]);
    v8f acc = {};
    acc = __builtin_amdgcn_wmma_f32_16x16x32_f16(false, af, false, bf, (short)0, acc, false, false);
    float* Sh = S + (size_t)h * LSEQ * LSEQ;
#pragma unroll
    for (int v = 0; v < 8; ++v)
        Sh[(size_t)(tm + v + (g << 3)) * LSEQ + (tn + mn)] = acc[v] * scale;
}

// ---------------- build V^T f16: VT[bb][h][n][k] = qkv[bb,k, 512 + h*32 + n] ----------------
__global__ void k_vt_build(const float* __restrict__ qkv, half_t* __restrict__ VT)
{
    int i = blockIdx.x * blockDim.x + threadIdx.x;   // 16*8*32*512
    if (i >= BSZ * NHEADS * 32 * LSEQ) return;
    int k = i & (LSEQ - 1);
    int n = (i >> 9) & 31;
    int h = (i >> 14) & 7;
    int bb = i >> 17;
    VT[i] = (half_t)qkv[((size_t)bb * LSEQ + k) * 768 + 2 * HDIM + h * 32 + n];
}

// ---------------- attention: O = P @ V (per head, one batch), V^T f16 operand ----------------
__global__ void k_attn_av(const float* __restrict__ S, const half_t* __restrict__ VT,
                          float* __restrict__ Ob)
{
    int h = blockIdx.z;
    int tm = blockIdx.x << 4;                      // 32 m-tiles
    int lane = threadIdx.x, g = lane >> 4, mn = lane & 15;
    const float* Srow = S + (size_t)h * LSEQ * LSEQ + (size_t)(tm + mn) * LSEQ + 16 * g;
    const half_t* V0 = VT + ((size_t)h * 32 + mn) * LSEQ + 16 * g;      // cols mn, mn+16
    const half_t* V1 = V0 + (size_t)16 * LSEQ;
    v8f acc0 = {}, acc1 = {};
    for (int k0 = 0; k0 < LSEQ; k0 += 32) {
        v16h af = load_fragA(Srow, k0);
        v16h b0 = load_fragB_t(V0, k0);
        v16h b1 = load_fragB_t(V1, k0);
        acc0 = __builtin_amdgcn_wmma_f32_16x16x32_f16(false, af, false, b0, (short)0, acc0, false, false);
        acc1 = __builtin_amdgcn_wmma_f32_16x16x32_f16(false, af, false, b1, (short)0, acc1, false, false);
    }
    float* O = Ob + h * 32;
#pragma unroll
    for (int v = 0; v < 8; ++v) {
        int row = tm + v + (g << 3);
        O[(size_t)row * HDIM + mn] = acc0[v];
        O[(size_t)row * HDIM + mn + 16] = acc1[v];
    }
}

// ---------------- wave-per-row softmax (in place) ----------------
__global__ void k_softmax_row(float* __restrict__ S, int cols)
{
    size_t row = blockIdx.x;
    float* p = S + row * (size_t)cols;
    int lane = threadIdx.x;
    int per = cols >> 5;
    float mx = -1e30f;
    for (int i = 0; i < per; ++i) mx = fmaxf(mx, p[lane + i * 32]);
    for (int m = 16; m >= 1; m >>= 1) mx = fmaxf(mx, __shfl_xor(mx, m, 32));
    float s = 0.f;
    for (int i = 0; i < per; ++i) { float e = __expf(p[lane + i * 32] - mx); p[lane + i * 32] = e; s += e; }
    for (int m = 16; m >= 1; m >>= 1) s += __shfl_xor(s, m, 32);
    float inv = 1.f / s;
    for (int i = 0; i < per; ++i) p[lane + i * 32] *= inv;
}

// ---------------- wave-per-row LayerNorm over 256 cols ----------------
__global__ void k_layernorm(const float* __restrict__ in, const float* __restrict__ gam,
                            const float* __restrict__ bet, float* __restrict__ out, int act)
{
    size_t row = blockIdx.x;
    int lane = threadIdx.x;
    const float* p = in + row * HDIM;
    float v[8]; float s = 0.f;
#pragma unroll
    for (int i = 0; i < 8; ++i) { v[i] = p[lane + i * 32]; s += v[i]; }
#pragma unroll
    for (int m = 16; m >= 1; m >>= 1) s += __shfl_xor(s, m, 32);
    float mean = s * (1.f / HDIM);
    float var = 0.f;
#pragma unroll
    for (int i = 0; i < 8; ++i) { float d = v[i] - mean; var += d * d; }
#pragma unroll
    for (int m = 16; m >= 1; m >>= 1) var += __shfl_xor(var, m, 32);
    float rstd = rsqrtf(var * (1.f / HDIM) + 1e-5f);
    float* q = out + row * HDIM;
#pragma unroll
    for (int i = 0; i < 8; ++i) {
        int c = lane + i * 32;
        float o = (v[i] - mean) * rstd * gam[c] + bet[c];
        if (act == 1) o = geluf(o);
        q[c] = o;
    }
}

// ---------------- depthwise causal conv (DCONV=4) + silu ----------------
__global__ void k_conv_silu(const float* __restrict__ big, const float* __restrict__ cw,
                            const float* __restrict__ cb, float* __restrict__ xc)
{
    int idx = blockIdx.x * blockDim.x + threadIdx.x;
    if (idx >= TOK * DIN) return;
    int d = idx & (DIN - 1);
    int t = idx >> 9;
    int l = t & (LSEQ - 1);
    float s = cb[d];
#pragma unroll
    for (int j = 0; j < 4; ++j) {
        int ls = l - 3 + j;
        if (ls >= 0) s += big[(size_t)(t - l + ls) * (2 * DIN) + d] * cw[d * 4 + j];
    }
    xc[idx] = siluf(s);
}

// ---------------- selective-scan: 16 lanes (one per state n) per (b,d) channel ----------------
__global__ void k_ssm_scan(const float* __restrict__ xc, const float* __restrict__ xp,
                           const float* __restrict__ wdt, const float* __restrict__ bdt,
                           const float* __restrict__ Am, const float* __restrict__ Dv,
                           float* __restrict__ ys)
{
    int grp = blockIdx.x * (blockDim.x >> 4) + (threadIdx.x >> 4);   // b*DIN + d
    int n = threadIdx.x & 15;
    int d = grp & (DIN - 1);
    int bb = grp >> 9;
    float An = -__expf(Am[d * DSTATE + n]);
    float w = wdt[d], bd = bdt[d], Dd = Dv[d];
    float h = 0.f;
    for (int l = 0; l < LSEQ; ++l) {
        int t = bb * LSEQ + l;
        const float* xpr = xp + (size_t)t * XPAD;
        float delta = softplusf_(xpr[0] * w + bd);
        float xv = xc[(size_t)t * DIN + d];
        h = __expf(delta * An) * h + delta * xpr[1 + n] * xv;
        float p = h * xpr[1 + DSTATE + n];
        p += __shfl_xor(p, 8, 16);
        p += __shfl_xor(p, 4, 16);
        p += __shfl_xor(p, 2, 16);
        p += __shfl_xor(p, 1, 16);
        if (n == 0) ys[(size_t)t * DIN + d] = p + xv * Dd;
    }
}

// ---------------- elementwise kernels ----------------
__global__ void k_mul(const float* a, const float* b, float* o, int n)
{ int i = blockIdx.x * blockDim.x + threadIdx.x; if (i < n) o[i] = a[i] * b[i]; }

__global__ void k_add(const float* a, const float* b, float* o, int n)
{ int i = blockIdx.x * blockDim.x + threadIdx.x; if (i < n) o[i] = a[i] + b[i]; }

__global__ void k_mul_silu_z(float* ys, const float* big)   // ys *= silu(z), z = big cols [512,1024)
{
    int i = blockIdx.x * blockDim.x + threadIdx.x;
    if (i >= TOK * DIN) return;
    int t = i >> 9, c = i & (DIN - 1);
    ys[i] *= siluf(big[(size_t)t * (2 * DIN) + DIN + c]);
}

__global__ void k_concat(const float* a, const float* b, float* o)   // -> [TOK,512]
{
    int i = blockIdx.x * blockDim.x + threadIdx.x;
    if (i >= TOK * 512) return;
    int t = i >> 9, c = i & 511;
    o[i] = (c < HDIM) ? a[(size_t)t * HDIM + c] : b[(size_t)t * HDIM + c - HDIM];
}

__global__ void k_combine(const float* g, const float* p, const float* hc, float* o, int n)
{ int i = blockIdx.x * blockDim.x + threadIdx.x; if (i < n) o[i] = g[i] * p[i] + (1.f - g[i]) * hc[i]; }

__global__ void k_pad_wxp(const float* __restrict__ w, float* __restrict__ wp)  // [512,33] -> [512,48]
{
    int i = blockIdx.x * blockDim.x + threadIdx.x;
    if (i >= DIN * XPAD) return;
    int r = i / XPAD, c = i - r * XPAD;
    wp[i] = (c < 2 * DSTATE + 1) ? w[r * (2 * DSTATE + 1) + c] : 0.f;
}

__global__ void k_pool_score(const float* ho, const float* w, const float* b0, float* s)
{
    int t = blockIdx.x * blockDim.x + threadIdx.x;
    if (t >= TOK) return;
    float acc = b0[0];
    const float* r = ho + (size_t)t * HDIM;
    for (int d = 0; d < HDIM; ++d) acc += r[d] * w[d];
    s[t] = acc;
}

__global__ void k_ctx(const float* a, const float* ho, float* ctx)
{
    int i = blockIdx.x * blockDim.x + threadIdx.x;
    if (i >= BSZ * HDIM) return;
    int b = i >> 8, d = i & (HDIM - 1);
    float s = 0.f;
    for (int l = 0; l < LSEQ; ++l)
        s += a[b * LSEQ + l] * ho[((size_t)b * LSEQ + l) * HDIM + d];
    ctx[i] = s;
}

__global__ void k_heads(const float* ctx, const float* wl, const float* bl,
                        const float* we, const float* be, const float* wa, const float* ba,
                        float* out)
{
    int b = threadIdx.x;
    if (b >= BSZ) return;
    const float* c = ctx + b * HDIM;
    for (int j = 0; j < 3; ++j) {
        float s = bl[j];
        for (int d = 0; d < HDIM; ++d) s += c[d] * wl[d * 3 + j];
        out[b * 3 + j] = s;
    }
    float e = be[0], a = ba[0];
    for (int d = 0; d < HDIM; ++d) { e += c[d] * we[d]; a += c[d] * wa[d]; }
    out[BSZ * 3 + b] = 0.5f * sigf(e) + 0.5f * sigf(a);
}

// ---------------- host orchestration ----------------
extern "C" void kernel_launch(void* const* d_in, const int* in_sizes, int n_in,
                              void* d_out, int out_size, void* d_ws, size_t ws_size,
                              hipStream_t stream)
{
    (void)in_sizes; (void)n_in; (void)out_size; (void)ws_size;
    // Flattened input order (dict insertion order of setup_inputs):
    // 0:x 1:w_inp 2:b_inp 3:in_g 4:in_b
    // mamba j (base 5+11j): ng nb w_in conv_w conv_b w_xp w_dt b_dt A D w_out
    // tblock j (base 38+11j): n1g n1b wqkv bqkv wo bo n2g n2b w_gate w_up w_down
    // cross (71..74): wqkv bqkv wo bo
    // 75:w_gate 76:b_gate 77:w_proj 78:b_proj 79:out_g 80:out_b 81:w_pool 82:b_pool
    // 83:w_logits 84:b_logits 85:w_econf 86:b_econf 87:w_aconf 88:b_aconf
    const float* x     = (const float*)d_in[0];
    const float* w_inp = (const float*)d_in[1];
    const float* b_inp = (const float*)d_in[2];
    const float* in_g  = (const float*)d_in[3];
    const float* in_b  = (const float*)d_in[4];

    float* ws = (float*)d_ws;
    size_t off = 0;
    auto alloc = [&](size_t n) { float* p = ws + off; off += n; return p; };
    float* H    = alloc((size_t)TOK * HDIM);
    float* HM   = alloc((size_t)TOK * HDIM);
    float* HT   = alloc((size_t)TOK * HDIM);
    float* XN   = alloc((size_t)TOK * HDIM);
    float* BIG  = alloc((size_t)TOK * (2 * DIN));   // xi|z / qkv / ffn gate
    float* BIG2 = alloc((size_t)TOK * DFF);         // ffn up
    float* XC   = alloc((size_t)TOK * DIN);         // conv out / hcat
    float* XP   = alloc((size_t)TOK * XPAD);
    float* YS   = alloc((size_t)TOK * DIN);
    float* SC   = alloc((size_t)NHEADS * LSEQ * LSEQ);  // per-batch score buffer
    float* AT   = alloc((size_t)TOK * HDIM);
    float* HC   = alloc((size_t)TOK * HDIM);
    float* GB   = alloc((size_t)TOK * HDIM);
    float* PB   = alloc((size_t)TOK * HDIM);
    float* WXPP = alloc((size_t)DIN * XPAD);                // padded w_xp (f32)
    half_t* WT1 = (half_t*)alloc((size_t)DFF * HDIM / 2);   // 262144 halfs
    half_t* WT2 = (half_t*)alloc((size_t)DFF * HDIM / 2);
    half_t* VTH = (half_t*)alloc((size_t)BSZ * NHEADS * 32 * LSEQ / 2);  // 2M halfs
    float* PS   = alloc((size_t)TOK);
    float* CTX  = alloc((size_t)BSZ * HDIM);

    const int EB = 256;  // elementwise block

    // transpose W[K,N] f32 -> WTbuf[N,K] f16, then WMMA GEMM
    auto gemm = [&](const float* A, const float* W, half_t* WTbuf,
                    const float* bias, const float* res, float* C,
                    int M, int N, int K, int lda, int ldc, int act) {
        k_transpose_f16<<<dim3((K * N + EB - 1) / EB), dim3(EB), 0, stream>>>(W, WTbuf, K, N);
        if ((N & 63) == 0) {
            int ntg = N >> 6;
            k_gemm<4><<<dim3(ntg * (M >> 4)), dim3(32), 0, stream>>>(A, WTbuf, bias, res, C,
                                                                     M, N, K, lda, ldc, act, ntg);
        } else {
            int ntg = N >> 4;
            k_gemm<1><<<dim3(ntg * (M >> 4)), dim3(32), 0, stream>>>(A, WTbuf, bias, res, C,
                                                                     M, N, K, lda, ldc, act, ntg);
        }
    };

    auto attention = [&](const float* qkv, float* outp) {
        const float scale = 0.17677669529663689f;   // 1/sqrt(32)
        k_vt_build<<<dim3((BSZ * NHEADS * 32 * LSEQ + EB - 1) / EB), dim3(EB), 0, stream>>>(qkv, VTH);
        for (int bb = 0; bb < BSZ; ++bb) {
            const float* base = qkv + (size_t)bb * LSEQ * 768;
            k_attn_scores<<<dim3(1024, 1, NHEADS), dim3(32), 0, stream>>>(base, base + HDIM, SC, scale);
            k_softmax_row<<<dim3(NHEADS * LSEQ), dim3(32), 0, stream>>>(SC, LSEQ);
            k_attn_av<<<dim3(32, 1, NHEADS), dim3(32), 0, stream>>>(
                SC, VTH + (size_t)bb * NHEADS * 32 * LSEQ, outp + (size_t)bb * LSEQ * HDIM);
        }
    };

    // ---- input projection + LN + gelu ----
    gemm(x, w_inp, WT1, b_inp, nullptr, XN, TOK, HDIM, 64, 64, HDIM, 0);
    k_layernorm<<<dim3(TOK), dim3(32), 0, stream>>>(XN, in_g, in_b, H, 1);

    // ---- mamba chain ----
    for (int j = 0; j < 3; ++j) {
        int base = 5 + 11 * j;
        const float* ng   = (const float*)d_in[base + 0];
        const float* nb   = (const float*)d_in[base + 1];
        const float* w_in = (const float*)d_in[base + 2];
        const float* cw   = (const float*)d_in[base + 3];
        const float* cb   = (const float*)d_in[base + 4];
        const float* wxp  = (const float*)d_in[base + 5];
        const float* wdt  = (const float*)d_in[base + 6];
        const float* bdt  = (const float*)d_in[base + 7];
        const float* Am   = (const float*)d_in[base + 8];
        const float* Dv   = (const float*)d_in[base + 9];
        const float* wout = (const float*)d_in[base + 10];
        const float* in = (j == 0) ? H : HM;
        k_layernorm<<<dim3(TOK), dim3(32), 0, stream>>>(in, ng, nb, XN, 0);
        gemm(XN, w_in, WT1, nullptr, nullptr, BIG, TOK, 2 * DIN, HDIM, HDIM, 2 * DIN, 0);
        k_conv_silu<<<dim3((TOK * DIN + EB - 1) / EB), dim3(EB), 0, stream>>>(BIG, cw, cb, XC);
        k_pad_wxp<<<dim3((DIN * XPAD + EB - 1) / EB), dim3(EB), 0, stream>>>(wxp, WXPP);
        gemm(XC, WXPP, WT2, nullptr, nullptr, XP, TOK, XPAD, DIN, DIN, XPAD, 0);
        k_ssm_scan<<<dim3(2048), dim3(64), 0, stream>>>(XC, XP, wdt, bdt, Am, Dv, YS);
        k_mul_silu_z<<<dim3((TOK * DIN + EB - 1) / EB), dim3(EB), 0, stream>>>(YS, BIG);
        gemm(YS, wout, WT1, nullptr, in, HM, TOK, HDIM, DIN, DIN, HDIM, 0);
    }

    // ---- transformer chain ----
    for (int j = 0; j < 3; ++j) {
        int base = 38 + 11 * j;
        const float* n1g  = (const float*)d_in[base + 0];
        const float* n1b  = (const float*)d_in[base + 1];
        const float* wqkv = (const float*)d_in[base + 2];
        const float* bqkv = (const float*)d_in[base + 3];
        const float* wo   = (const float*)d_in[base + 4];
        const float* bo   = (const float*)d_in[base + 5];
        const float* n2g  = (const float*)d_in[base + 6];
        const float* n2b  = (const float*)d_in[base + 7];
        const float* wg   = (const float*)d_in[base + 8];
        const float* wu   = (const float*)d_in[base + 9];
        const float* wd   = (const float*)d_in[base + 10];
        const float* in = (j == 0) ? H : HT;
        k_layernorm<<<dim3(TOK), dim3(32), 0, stream>>>(in, n1g, n1b, XN, 0);
        gemm(XN, wqkv, WT1, bqkv, nullptr, BIG, TOK, 3 * HDIM, HDIM, HDIM, 3 * HDIM, 0);
        attention(BIG, AT);
        gemm(AT, wo, WT1, bo, in, HT, TOK, HDIM, HDIM, HDIM, HDIM, 0);
        k_layernorm<<<dim3(TOK), dim3(32), 0, stream>>>(HT, n2g, n2b, XN, 0);
        gemm(XN, wg, WT1, nullptr, nullptr, BIG, TOK, DFF, HDIM, HDIM, DFF, 2);
        gemm(XN, wu, WT2, nullptr, nullptr, BIG2, TOK, DFF, HDIM, HDIM, DFF, 0);
        k_mul<<<dim3((TOK * DFF + EB - 1) / EB), dim3(EB), 0, stream>>>(BIG, BIG2, BIG, TOK * DFF);
        gemm(BIG, wd, WT1, nullptr, HT, HT, TOK, HDIM, DFF, DFF, HDIM, 0);
    }

    // ---- cross attention on hm + ht ----
    k_add<<<dim3((TOK * HDIM + EB - 1) / EB), dim3(EB), 0, stream>>>(HM, HT, H, TOK * HDIM);
    {
        const float* cwqkv = (const float*)d_in[71];
        const float* cbqkv = (const float*)d_in[72];
        const float* cwo   = (const float*)d_in[73];
        const float* cbo   = (const float*)d_in[74];
        gemm(H, cwqkv, WT1, cbqkv, nullptr, BIG, TOK, 3 * HDIM, HDIM, HDIM, 3 * HDIM, 0);
        attention(BIG, AT);
        gemm(AT, cwo, WT1, cbo, nullptr, HC, TOK, HDIM, HDIM, HDIM, HDIM, 0);
    }

    // ---- gated fusion + output head ----
    const float* w_gate = (const float*)d_in[75];
    const float* b_gate = (const float*)d_in[76];
    const float* w_proj = (const float*)d_in[77];
    const float* b_proj = (const float*)d_in[78];
    const float* out_g  = (const float*)d_in[79];
    const float* out_b  = (const float*)d_in[80];
    const float* w_pool = (const float*)d_in[81];
    const float* b_pool = (const float*)d_in[82];

    k_concat<<<dim3((TOK * 512 + EB - 1) / EB), dim3(EB), 0, stream>>>(HM, HT, XC);
    gemm(XC, w_gate, WT1, b_gate, nullptr, GB, TOK, HDIM, 2 * HDIM, 2 * HDIM, HDIM, 3);
    gemm(XC, w_proj, WT2, b_proj, nullptr, PB, TOK, HDIM, 2 * HDIM, 2 * HDIM, HDIM, 0);
    k_combine<<<dim3((TOK * HDIM + EB - 1) / EB), dim3(EB), 0, stream>>>(GB, PB, HC, XN, TOK * HDIM);
    k_layernorm<<<dim3(TOK), dim3(32), 0, stream>>>(XN, out_g, out_b, H, 0);
    k_pool_score<<<dim3((TOK + EB - 1) / EB), dim3(EB), 0, stream>>>(H, w_pool, b_pool, PS);
    k_softmax_row<<<dim3(BSZ), dim3(32), 0, stream>>>(PS, LSEQ);
    k_ctx<<<dim3((BSZ * HDIM + EB - 1) / EB), dim3(EB), 0, stream>>>(PS, H, CTX);
    k_heads<<<dim3(1), dim3(32), 0, stream>>>(CTX,
        (const float*)d_in[83], (const float*)d_in[84],
        (const float*)d_in[85], (const float*)d_in[86],
        (const float*)d_in[87], (const float*)d_in[88],
        (float*)d_out);
}
